// MCGraphNet_1262720385395
// MI455X (gfx1250) — compile-verified, matrix operand order checked by reading, not run on hardware
//
#include <hip/hip_runtime.h>
#include <hip/hip_bf16.h>

// ---------------------------------------------------------------------------
// MCGraphNet on MI455X (gfx1250): fused WMMA MLP blocks, LDS-staged tiles.
// One wave (32 lanes) computes a 16-row x 64-col MLP output tile via
// V_WMMA_F32_16X16X32_F16 (4 N-tiles, K/32 k-steps). Weights are pre-packed
// into the per-lane WMMA B-fragment layout by pack_kernel. Staging uses
// float2 loads + packed f16 converts + dword DS stores.
// ---------------------------------------------------------------------------

typedef __attribute__((ext_vector_type(16))) _Float16 v16h;
typedef __attribute__((ext_vector_type(8)))  float    v8f;
typedef __attribute__((ext_vector_type(4)))  unsigned v4u;

union HFrag { v16h h; v4u q[2]; };

__device__ __forceinline__ v16h load_a_frag(const _Float16* p) {
  // p points at lds row base + kb*32 halves + ((lane&16)?8:0); 16B aligned.
  HFrag f;
  f.q[0] = *(const v4u*)(p);        // halves for K = base+0..7
  f.q[1] = *(const v4u*)(p + 16);   // halves for K = base+16..23
  return f.h;
}

__device__ __forceinline__ v16h load_b_frag(const unsigned* w, int tile, int lane) {
  const v4u* p = (const v4u*)(w + ((size_t)tile * 32 + lane) * 8);
  HFrag f; f.q[0] = p[0]; f.q[1] = p[1]; return f.h;
}

#define WMMA_F16(a, b, c) \
  __builtin_amdgcn_wmma_f32_16x16x32_f16(false, (a), false, (b), (short)0, (c), false, false)

__device__ __forceinline__ void lds_fence() {
  __asm volatile("s_wait_dscnt 0x0" ::: "memory");
}

// pack two f32 -> two f16 in one dword (compiler fuses to v_cvt_pk_f16_f32)
__device__ __forceinline__ unsigned pack_h2(float a, float b) {
  _Float16 ha = (_Float16)a, hb = (_Float16)b;
  return (unsigned)__builtin_bit_cast(unsigned short, ha) |
         ((unsigned)__builtin_bit_cast(unsigned short, hb) << 16);
}

// order-preserving float<->uint transform for atomicMax on floats
__device__ __forceinline__ unsigned encf(float x) {
  unsigned u = __float_as_uint(x);
  return (u & 0x80000000u) ? ~u : (u | 0x80000000u);
}
__device__ __forceinline__ float decf(unsigned u) {
  unsigned b = (u & 0x80000000u) ? (u & 0x7fffffffu) : ~u;
  return __uint_as_float(b);
}

// --------------------------- GEMM layer helpers ----------------------------

// 16 rows x 64 cols, input K = KB*32 (f16, LDS, strideIn halves), ReLU,
// f16 output to LDS (stride 64).
template<int KB>
__device__ __forceinline__ void layer_fwd(const _Float16* ldsIn, int strideIn,
                                          const unsigned* wpk, const float* bias,
                                          _Float16* ldsOut, int lane) {
  v8f acc[4] = {};
  const _Float16* base = ldsIn + (lane & 15) * strideIn + ((lane & 16) ? 8 : 0);
#pragma unroll
  for (int kb = 0; kb < KB; ++kb) {
    v16h a = load_a_frag(base + kb * 32);
#pragma unroll
    for (int nt = 0; nt < 4; ++nt) {
      v16h b = load_b_frag(wpk, kb * 4 + nt, lane);
      acc[nt] = WMMA_F16(a, b, acc[nt]);
    }
  }
  const int n = lane & 15, mb = (lane & 16) ? 8 : 0;
#pragma unroll
  for (int nt = 0; nt < 4; ++nt) {
    float bv = bias[nt * 16 + n];
#pragma unroll
    for (int m = 0; m < 8; ++m) {
      float v = acc[nt][m] + bv;
      v = fmaxf(v, 0.0f);
      ldsOut[(mb + m) * 64 + nt * 16 + n] = (_Float16)v;
    }
  }
}

// Final layer of a block: bias + LayerNorm(64) across cols, f32 results in
// outv[nt][m] for row (mb+m), col (nt*16 + (lane&15)).
template<int KB>
__device__ __forceinline__ void layer_last_ln(const _Float16* ldsIn, int strideIn,
                                              const unsigned* wpk, const float* bias,
                                              const float* gamma, const float* beta,
                                              float (&outv)[4][8], int lane) {
  v8f acc[4] = {};
  const _Float16* base = ldsIn + (lane & 15) * strideIn + ((lane & 16) ? 8 : 0);
#pragma unroll
  for (int kb = 0; kb < KB; ++kb) {
    v16h a = load_a_frag(base + kb * 32);
#pragma unroll
    for (int nt = 0; nt < 4; ++nt) {
      v16h b = load_b_frag(wpk, kb * 4 + nt, lane);
      acc[nt] = WMMA_F16(a, b, acc[nt]);
    }
  }
  const int n = lane & 15;
  float vals[4][8];
#pragma unroll
  for (int nt = 0; nt < 4; ++nt) {
    float bv = bias[nt * 16 + n];
#pragma unroll
    for (int m = 0; m < 8; ++m) vals[nt][m] = acc[nt][m] + bv;
  }
  // Row m lives on the 16-lane half-wave: reduce sums across xor 1,2,4,8.
#pragma unroll
  for (int m = 0; m < 8; ++m) {
    float s = vals[0][m] + vals[1][m] + vals[2][m] + vals[3][m];
    float q = vals[0][m] * vals[0][m] + vals[1][m] * vals[1][m]
            + vals[2][m] * vals[2][m] + vals[3][m] * vals[3][m];
#pragma unroll
    for (int off = 1; off < 16; off <<= 1) {
      s += __shfl_xor(s, off, 32);
      q += __shfl_xor(q, off, 32);
    }
    float mean = s * (1.0f / 64.0f);
    float var  = q * (1.0f / 64.0f) - mean * mean;
    float rstd = rsqrtf(var + 1e-5f);
#pragma unroll
    for (int nt = 0; nt < 4; ++nt)
      outv[nt][m] = (vals[nt][m] - mean) * rstd * gamma[nt * 16 + n] + beta[nt * 16 + n];
  }
}

// ------------------------------- kernels -----------------------------------

// Pack W (din x dout, row-major f32) into per-lane WMMA B fragments (f16).
__global__ void pack_kernel(const float* __restrict__ W, unsigned* __restrict__ out,
                            int din, int dout, int KB, int NT) {
  int tid = blockIdx.x * blockDim.x + threadIdx.x;
  int total = KB * NT * 32;
  if (tid >= total) return;
  int lane = tid & 31, t = tid >> 5;
  int nt = t % NT, kb = t / NT;
  int n = nt * 16 + (lane & 15);
  int koff = (lane & 16) ? 8 : 0;
#pragma unroll
  for (int vg = 0; vg < 8; ++vg) {
    unsigned u = 0;
    for (int w = 0; w < 2; ++w) {
      int k = kb * 32 + (vg < 4 ? 2 * vg : 16 + 2 * (vg - 4)) + koff + w;
      float f = (k < din && n < dout) ? W[(size_t)k * dout + n] : 0.0f;
      _Float16 hf = (_Float16)f;
      unsigned short us = __builtin_bit_cast(unsigned short, hf);
      u |= ((unsigned)us) << (16 * w);
    }
    out[((size_t)t * 32 + lane) * 8 + vg] = u;
  }
}

__global__ void fill_kernel(unsigned* __restrict__ p, unsigned v, int n) {
  int i = blockIdx.x * blockDim.x + threadIdx.x;
  if (i < n) p[i] = v;
}

// Encoder block: rows x DIN (f32) -> MLP(DIN->64->64->64)+LN -> out (f32)
template<int DIN>
__global__ __launch_bounds__(128) void enc_kernel(
    const float* __restrict__ in, float* __restrict__ out,
    const unsigned* w1, const float* b1, const unsigned* w2, const float* b2,
    const unsigned* w3, const float* b3, const float* g, const float* be, int rows) {
  __shared__ __align__(16) _Float16 sIn[4][16 * 32];
  __shared__ __align__(16) _Float16 sB1[4][16 * 64];
  __shared__ __align__(16) _Float16 sB2[4][16 * 64];
  int lane = threadIdx.x & 31, wv = threadIdx.x >> 5;
  int row0 = blockIdx.x * 64 + wv * 16;
  if (row0 >= rows) return;                 // wave-uniform
  _Float16* sin = sIn[wv]; _Float16* s1 = sB1[wv]; _Float16* s2 = sB2[wv];
  // 16 rows x 16 dwords (32 halves, zero-padded past DIN)
#pragma unroll
  for (int i = lane; i < 256; i += 32) {
    int r = i >> 4, u = i & 15;
    int id = row0 + r; if (id > rows - 1) id = rows - 1;
    int c = 2 * u;
    float a = 0.0f, b = 0.0f;
    if (c + 1 < DIN) {
      float2 v = *(const float2*)(in + (size_t)id * DIN + c);
      a = v.x; b = v.y;
    } else if (c < DIN) {
      a = in[(size_t)id * DIN + c];
    }
    ((unsigned*)sin)[r * 16 + u] = pack_h2(a, b);
  }
  lds_fence();
  layer_fwd<1>(sin, 32, w1, b1, s1, lane);
  lds_fence();
  layer_fwd<2>(s1, 64, w2, b2, s2, lane);
  lds_fence();
  float outv[4][8];
  layer_last_ln<2>(s2, 64, w3, b3, g, be, outv, lane);
  int n = lane & 15, mb = (lane & 16) ? 8 : 0;
#pragma unroll
  for (int m = 0; m < 8; ++m) {
    int id = row0 + mb + m;
    if (id >= rows) continue;
#pragma unroll
    for (int nt = 0; nt < 4; ++nt)
      out[(size_t)id * 64 + nt * 16 + n] = outv[nt][m];
  }
}

// Edge block: gather concat(h[tgt], h[src], e) -> MLP(192->64->64->64)+LN,
// residual e += e_new (in-place), scatter-aggregate into agg.
// SYMM: process rows [0,E2); mirrored edges get -e_new; max-aggregation.
template<bool SYMM>
__global__ __launch_bounds__(128) void edge_kernel(
    const float* __restrict__ h, float* __restrict__ e,
    const int* __restrict__ srcI, const int* __restrict__ tgtI,
    const unsigned* w1, const float* b1, const unsigned* w2, const float* b2,
    const unsigned* w3, const float* b3, const float* g, const float* be,
    float* __restrict__ agg, int rowsE, int E2) {
  __shared__ __align__(16) _Float16 sIn[4][16 * 192];
  __shared__ __align__(16) _Float16 sB1[4][16 * 64];
  __shared__ __align__(16) _Float16 sB2[4][16 * 64];
  int lane = threadIdx.x & 31, wv = threadIdx.x >> 5;
  int row0 = blockIdx.x * 64 + wv * 16;
  if (row0 >= rowsE) return;                // wave-uniform
  _Float16* sin = sIn[wv]; _Float16* s1 = sB1[wv]; _Float16* s2 = sB2[wv];
#pragma unroll
  for (int r = 0; r < 16; ++r) {
    int id = row0 + r; if (id > rowsE - 1) id = rowsE - 1;
    int t = tgtI[id], s = srcI[id];
    float2 a = ((const float2*)(h + (size_t)t * 64))[lane];
    float2 b = ((const float2*)(h + (size_t)s * 64))[lane];
    float2 c = ((const float2*)(e + (size_t)id * 64))[lane];
    unsigned* row = (unsigned*)(sin + r * 192);
    row[lane]      = pack_h2(a.x, a.y);
    row[32 + lane] = pack_h2(b.x, b.y);
    row[64 + lane] = pack_h2(c.x, c.y);
  }
  lds_fence();
  layer_fwd<6>(sin, 192, w1, b1, s1, lane);
  lds_fence();
  layer_fwd<2>(s1, 64, w2, b2, s2, lane);
  lds_fence();
  float outv[4][8];
  layer_last_ln<2>(s2, 64, w3, b3, g, be, outv, lane);
  int n = lane & 15, mb = (lane & 16) ? 8 : 0;
  unsigned* aggU = (unsigned*)agg;
#pragma unroll
  for (int m = 0; m < 8; ++m) {
    int id = row0 + mb + m;
    if (id >= rowsE) continue;
    int t1 = tgtI[id];
    int t2 = SYMM ? tgtI[id + E2] : 0;
#pragma unroll
    for (int nt = 0; nt < 4; ++nt) {
      int col = nt * 16 + n;
      float v = outv[nt][m];
      size_t off = (size_t)id * 64 + col;
      if (!SYMM) {
        e[off] += v;
        atomicAdd(&agg[(size_t)t1 * 64 + col], v);
      } else {
        e[off] += v;
        size_t off2 = (size_t)(id + E2) * 64 + col;
        e[off2] -= v;  // mirrored edge residual: e + (-e_new)
        atomicMax(&aggU[(size_t)t1 * 64 + col], encf(v));
        atomicMax(&aggU[(size_t)t2 * 64 + col], encf(-v));
      }
    }
  }
}

// Node block: concat(agg, h) -> MLP(128->64->64->64)+LN, residual h += out.
template<bool MAXM>
__global__ __launch_bounds__(128) void node_kernel(
    float* __restrict__ hio, const float* __restrict__ agg,
    const unsigned* w1, const float* b1, const unsigned* w2, const float* b2,
    const unsigned* w3, const float* b3, const float* g, const float* be, int N) {
  __shared__ __align__(16) _Float16 sIn[4][16 * 128];
  __shared__ __align__(16) _Float16 sB1[4][16 * 64];
  __shared__ __align__(16) _Float16 sB2[4][16 * 64];
  int lane = threadIdx.x & 31, wv = threadIdx.x >> 5;
  int row0 = blockIdx.x * 64 + wv * 16;
  if (row0 >= N) return;                    // wave-uniform
  _Float16* sin = sIn[wv]; _Float16* s1 = sB1[wv]; _Float16* s2 = sB2[wv];
  const unsigned* aggU = (const unsigned*)agg;
#pragma unroll
  for (int r = 0; r < 16; ++r) {
    int id = row0 + r; if (id > N - 1) id = N - 1;
    float ax, ay;
    if (MAXM) {
      uint2 u = ((const uint2*)(aggU + (size_t)id * 64))[lane];
      float fx = decf(u.x), fy = decf(u.y);
      ax = (fx < -3.0e38f || fx != fx) ? 0.0f : fx;   // empty segment -> 0
      ay = (fy < -3.0e38f || fy != fy) ? 0.0f : fy;
    } else {
      float2 a = ((const float2*)(agg + (size_t)id * 64))[lane];
      ax = a.x; ay = a.y;
    }
    float2 hh = ((const float2*)(hio + (size_t)id * 64))[lane];
    unsigned* row = (unsigned*)(sin + r * 128);
    row[lane]      = pack_h2(ax, ay);
    row[32 + lane] = pack_h2(hh.x, hh.y);
  }
  lds_fence();
  layer_fwd<4>(sin, 128, w1, b1, s1, lane);
  lds_fence();
  layer_fwd<2>(s1, 64, w2, b2, s2, lane);
  lds_fence();
  float outv[4][8];
  layer_last_ln<2>(s2, 64, w3, b3, g, be, outv, lane);
  int n = lane & 15, mb = (lane & 16) ? 8 : 0;
#pragma unroll
  for (int m = 0; m < 8; ++m) {
    int id = row0 + mb + m;
    if (id >= N) continue;
#pragma unroll
    for (int nt = 0; nt < 4; ++nt)
      hio[(size_t)id * 64 + nt * 16 + n] += outv[nt][m];
  }
}

// Decoder: MLP(64->64->64->3), no LN. Output f32 N x 3.
__global__ __launch_bounds__(128) void dec_kernel(
    const float* __restrict__ h, float* __restrict__ out,
    const unsigned* w1, const float* b1, const unsigned* w2, const float* b2,
    const unsigned* w3, const float* b3, int N) {
  __shared__ __align__(16) _Float16 sIn[4][16 * 64];
  __shared__ __align__(16) _Float16 sB1[4][16 * 64];
  __shared__ __align__(16) _Float16 sB2[4][16 * 64];
  int lane = threadIdx.x & 31, wv = threadIdx.x >> 5;
  int row0 = blockIdx.x * 64 + wv * 16;
  if (row0 >= N) return;                    // wave-uniform
  _Float16* sin = sIn[wv]; _Float16* s1 = sB1[wv]; _Float16* s2 = sB2[wv];
#pragma unroll
  for (int r = 0; r < 16; ++r) {
    int id = row0 + r; if (id > N - 1) id = N - 1;
    float2 a = ((const float2*)(h + (size_t)id * 64))[lane];
    ((unsigned*)(sin + r * 64))[lane] = pack_h2(a.x, a.y);
  }
  lds_fence();
  layer_fwd<2>(sin, 64, w1, b1, s1, lane);
  lds_fence();
  layer_fwd<2>(s1, 64, w2, b2, s2, lane);
  lds_fence();
  // last layer: dout=3 -> single N-tile (NT=1)
  v8f acc = {};
  const _Float16* base = s2 + (lane & 15) * 64 + ((lane & 16) ? 8 : 0);
#pragma unroll
  for (int kb = 0; kb < 2; ++kb) {
    v16h a = load_a_frag(base + kb * 32);
    v16h b = load_b_frag(w3, kb, lane);
    acc = WMMA_F16(a, b, acc);
  }
  int n = lane & 15, mb = (lane & 16) ? 8 : 0;
  if (n < 3) {
    float bv = b3[n];
#pragma unroll
    for (int m = 0; m < 8; ++m) {
      int id = row0 + mb + m;
      if (id < N) out[(size_t)id * 3 + n] = acc[m] + bv;
    }
  }
}

// ------------------------------- launcher ----------------------------------

extern "C" void kernel_launch(void* const* d_in, const int* in_sizes, int n_in,
                              void* d_out, int out_size, void* d_ws, size_t ws_size,
                              hipStream_t stream) {
  (void)n_in; (void)out_size; (void)ws_size;

  const int N  = in_sizes[0] / 30;   // x: (N, 30)
  const int E  = in_sizes[1] / 4;    // edge_attr: (E, 4)
  const int E2 = E / 2;

  const float* x   = (const float*)d_in[0];
  const float* ea  = (const float*)d_in[1];
  const int*   ei  = (const int*)d_in[2];
  const int* srcI  = ei;             // edge_index[0]
  const int* tgtI  = ei + E;         // edge_index[1]

  // -------- param leaves in setup_inputs() dict (insertion) order ----------
  int cur = 3;
  auto F = [&]() { return (const float*)d_in[cur++]; };
  struct BP { const float *W0,*b0,*W1,*b1,*W2,*b2,*g,*be; };
  auto readBlk = [&](BP& p) {
    p.W0=F(); p.b0=F(); p.W1=F(); p.b1=F(); p.W2=F(); p.b2=F(); p.g=F(); p.be=F();
  };
  BP encN, encE, inE[4], inN[4], symE, symN;
  readBlk(encN); readBlk(encE);
  for (int i = 0; i < 4; ++i) { readBlk(inE[i]); readBlk(inN[i]); }
  readBlk(symE); readBlk(symN);
  struct MP { const float *W0,*b0,*W1,*b1,*W2,*b2; } dec;
  dec.W0=F(); dec.b0=F(); dec.W1=F(); dec.b1=F(); dec.W2=F(); dec.b2=F();

  // ------------------------ workspace carving ------------------------------
  char* ws = (char*)d_ws;
  size_t off = 0;
  auto carve = [&](size_t bytes) -> void* {
    void* p = ws + off;
    off = (off + bytes + 255) & ~(size_t)255;
    return p;
  };
  auto doPack = [&](const float* W, int din, int dout, int KB, int NT) -> unsigned* {
    unsigned* o = (unsigned*)carve((size_t)KB * NT * 32 * 8 * 4);
    int total = KB * NT * 32;
    pack_kernel<<<(total + 127) / 128, 128, 0, stream>>>(W, o, din, dout, KB, NT);
    return o;
  };
  struct PK { unsigned *w0, *w1, *w2; };
  auto packBlk = [&](const BP& b, int d0, int KB0) -> PK {
    PK p;
    p.w0 = doPack(b.W0, d0, 64, KB0, 4);
    p.w1 = doPack(b.W1, 64, 64, 2, 4);
    p.w2 = doPack(b.W2, 64, 64, 2, 4);
    return p;
  };

  PK pEncN = packBlk(encN, 30, 1);
  PK pEncE = packBlk(encE, 4, 1);
  PK pInE[4], pInN[4];
  for (int i = 0; i < 4; ++i) {
    pInE[i] = packBlk(inE[i], 192, 6);
    pInN[i] = packBlk(inN[i], 128, 4);
  }
  PK pSymE = packBlk(symE, 192, 6);
  PK pSymN = packBlk(symN, 128, 4);
  unsigned* pDec0 = doPack(dec.W0, 64, 64, 2, 4);
  unsigned* pDec1 = doPack(dec.W1, 64, 64, 2, 4);
  unsigned* pDec2 = doPack(dec.W2, 64, 3, 2, 1);

  float* hbuf = (float*)carve((size_t)N * 64 * 4);
  float* ebuf = (float*)carve((size_t)E * 64 * 4);
  float* agg  = (float*)carve((size_t)N * 64 * 4);

  const int nBlkN  = (N + 63) / 64;
  const int nBlkE  = (E + 63) / 64;
  const int nBlkE2 = (E2 + 63) / 64;
  const int fillN  = N * 64;
  const int fillB  = (fillN + 255) / 256;

  // ------------------------------ pipeline ---------------------------------
  enc_kernel<30><<<nBlkN, 128, 0, stream>>>(
      x, hbuf, pEncN.w0, encN.b0, pEncN.w1, encN.b1, pEncN.w2, encN.b2, encN.g, encN.be, N);
  enc_kernel<4><<<nBlkE, 128, 0, stream>>>(
      ea, ebuf, pEncE.w0, encE.b0, pEncE.w1, encE.b1, pEncE.w2, encE.b2, encE.g, encE.be, E);

  for (int i = 0; i < 4; ++i) {
    fill_kernel<<<fillB, 256, 0, stream>>>((unsigned*)agg, 0u, fillN);
    edge_kernel<false><<<nBlkE, 128, 0, stream>>>(
        hbuf, ebuf, srcI, tgtI,
        pInE[i].w0, inE[i].b0, pInE[i].w1, inE[i].b1, pInE[i].w2, inE[i].b2,
        inE[i].g, inE[i].be, agg, E, E2);
    node_kernel<false><<<nBlkN, 128, 0, stream>>>(
        hbuf, agg,
        pInN[i].w0, inN[i].b0, pInN[i].w1, inN[i].b1, pInN[i].w2, inN[i].b2,
        inN[i].g, inN[i].be, N);
  }

  // symmetric step: max aggregation (agg init = enc(-inf) = 0x007FFFFF)
  fill_kernel<<<fillB, 256, 0, stream>>>((unsigned*)agg, 0x007FFFFFu, fillN);
  edge_kernel<true><<<nBlkE2, 128, 0, stream>>>(
      hbuf, ebuf, srcI, tgtI,
      pSymE.w0, symE.b0, pSymE.w1, symE.b1, pSymE.w2, symE.b2,
      symE.g, symE.be, agg, E2, E2);
  node_kernel<true><<<nBlkN, 128, 0, stream>>>(
      hbuf, agg,
      pSymN.w0, symN.b0, pSymN.w1, symN.b1, pSymN.w2, symN.b2,
      symN.g, symN.be, N);

  dec_kernel<<<nBlkN, 128, 0, stream>>>(
      hbuf, (float*)d_out, pDec0, dec.b0, pDec1, dec.b1, pDec2, dec.b2, N);
}
